// LocalInjectionAttnProcessor_34265249088251
// MI455X (gfx1250) — matrix operand order repaired and checked by reading, not compiled, not run on hardware
//
#include <hip/hip_runtime.h>
#include <cstdint>
#include <cstddef>

typedef __bf16 bf16;
typedef __attribute__((ext_vector_type(16))) __bf16 v16bf;
typedef __attribute__((ext_vector_type(4)))  __bf16 v4bf;
typedef __attribute__((ext_vector_type(8)))  float  v8f;
typedef __attribute__((ext_vector_type(4)))  unsigned int u32x4;

// Problem constants (fixed by the reference)
#define Bb    4
#define SQL   2048
#define SKVL  2048
#define DD    1024
#define RR    4
#define HH    16
#define HDIM  64

// ---------------------------------------------------------------------------
// Fragment loaders: 2 x 16B LDS/global reads (ds_load_b128 / global_load_b128)
// ---------------------------------------------------------------------------
union FragU { v16bf v; u32x4 q[2]; };

// A-style fragment: two 8-half runs at p and p+16 halves (ISA 7.12.2 A layout)
__device__ inline v16bf ld_frag_a(const bf16* p) {
    FragU f;
    f.q[0] = *(const u32x4*)p;
    f.q[1] = *(const u32x4*)(p + 16);
    return f.v;
}
// Contiguous fragment: 16 consecutive halves (for K-major staged tiles)
__device__ inline v16bf ld_frag_c(const bf16* p) {
    FragU f;
    f.q[0] = *(const u32x4*)p;
    f.q[1] = *(const u32x4*)(p + 8);
    return f.v;
}

// Flat pointer -> LDS byte offset (LDS aperture keeps offset in addr[31:0])
__device__ inline uint32_t lds_off(const void* p) {
    return (uint32_t)(uintptr_t)p;
}

// Async DMA of one 16B granule from global to LDS (tracked by ASYNCcnt)
__device__ inline void async_b128(uint32_t loff, uint32_t voff, uint64_t sbase) {
    asm volatile("global_load_async_to_lds_b128 %0, %1, %2"
                 :: "v"(loff), "v"(voff), "s"(sbase) : "memory");
}
__device__ inline void wait_async0() {
    asm volatile("s_wait_asynccnt 0" ::: "memory");
}

// ---------------------------------------------------------------------------
// fp32 -> bf16 conversion (float4 in, 64-bit packed bf16 out)
// ---------------------------------------------------------------------------
__global__ __launch_bounds__(256)
void cvt_f32_bf16(const float* __restrict__ in, bf16* __restrict__ out, int n) {
    int i = (blockIdx.x * 256 + threadIdx.x) * 4;
    if (i + 3 < n) {
        float4 f = *(const float4*)(in + i);
        v4bf o;
        o[0] = (bf16)f.x; o[1] = (bf16)f.y; o[2] = (bf16)f.z; o[3] = (bf16)f.w;
        *(v4bf*)(out + i) = o;
    } else {
        for (int j = 0; j < 4; ++j)
            if (i + j < n) out[i + j] = (bf16)in[i + j];
    }
}

// ---------------------------------------------------------------------------
// LoRA down projection: t[M,4] = Xbf16[M,K] @ down[K,4]   (rank 4, tiny)
// ---------------------------------------------------------------------------
__global__ __launch_bounds__(256)
void lora_down_kernel(const bf16* __restrict__ X, const float* __restrict__ down,
                      float* __restrict__ t, int M, int Kd) {
    int idx = blockIdx.x * 256 + threadIdx.x;     // over M*4
    if (idx >= M * 4) return;
    int row = idx >> 2;
    int r   = idx & 3;
    const bf16* xp = X + (size_t)row * Kd;
    float acc = 0.f;
    for (int k = 0; k < Kd; k += 16) {
        v16bf xv = ld_frag_c(xp + k);            // 16 contiguous halves (b128 x2)
        #pragma unroll
        for (int j = 0; j < 16; ++j)
            acc += (float)xv[j] * down[(k + j) * 4 + r];
    }
    t[idx] = acc;
}

// ---------------------------------------------------------------------------
// WMMA bf16 GEMM:  out = scale * (A[M,K] @ B[K,N] + t @ up) (+ bias)
// Block tile 128x128, K-step 32, 256 threads = 8 waves (4x2 wave grid),
// wave tile 32x64 = 2x4 WMMA 16x16 accumulators.
// Double-buffered LDS; A tile staged via global_load_async_to_lds_b128;
// B tile staged transposed (column-major) for contiguous 128-bit frag reads.
// SCALED is used by the Q projection to fold 1/sqrt(hd)=0.125 (a power of
// two -> exact in bf16) into Q, removing per-chunk scaling from attention.
// ---------------------------------------------------------------------------
#define BM 128
#define BN 128
#define BK 32
#define LDA_S 40     // halves per A row (32 data + 8 pad) = 80B, 16B aligned
#define LDB_T 40     // halves per B column (32 data + 8 pad) = 80B, 16B aligned

template<bool HAS_BIAS, bool OUT_F32, bool SCALED>
__global__ __launch_bounds__(256)
void gemm_bf16_wmma(const bf16* __restrict__ A, const bf16* __restrict__ Bw,
                    const float* __restrict__ lora_t,   // [M,4]
                    const float* __restrict__ lora_up,  // [4,N]
                    const float* __restrict__ bias,     // [N] (if HAS_BIAS)
                    bf16* __restrict__ out_bf,          // (if !OUT_F32)
                    float* __restrict__ out_f,          // (if OUT_F32)
                    float out_scale,
                    int M, int N, int Kd)
{
    __shared__ bf16 As[2][BM * LDA_S];   // row-major rows of A tile
    __shared__ bf16 Bt[2][BN * LDB_T];   // column-major (transposed) B tile

    const int tid  = threadIdx.x;
    const int lane = tid & 31;
    const int wave = tid >> 5;
    const int wm = wave >> 1;            // 0..3 -> 32-row strip
    const int wn = wave & 1;             // 0..1 -> 64-col strip
    const int m_wave = wm * 32;
    const int n_wave = wn * 64;
    const int m_blk = blockIdx.y * BM;
    const int n_blk = blockIdx.x * BN;
    const int half = lane >> 4;
    const int cn = lane & 15;

    v8f acc[2][4];
    #pragma unroll
    for (int i = 0; i < 2; ++i)
        #pragma unroll
        for (int t = 0; t < 4; ++t)
            #pragma unroll
            for (int j = 0; j < 8; ++j) acc[i][t][j] = 0.f;

    // Stage tile at K-offset kk into LDS buffer sb
    auto stageAB = [&](int sb, int kk) {
        // A: 128 rows x 4 granules of 16B via async DMA
        uint64_t abase = (uint64_t)(uintptr_t)(A + (size_t)m_blk * Kd + kk);
        uint32_t asB = lds_off(&As[sb][0]);
        for (int i = tid; i < BM * 4; i += 256) {
            uint32_t row = (uint32_t)i >> 2, g = (uint32_t)i & 3;
            async_b128(asB + row * (LDA_S * 2) + g * 16,
                       row * (uint32_t)(Kd * 2) + g * 16, abase);
        }
        // B: load dword pairs (k,k+1) x colpair, store transposed
        uint32_t* bt = (uint32_t*)&Bt[sb][0];
        for (int i = tid; i < (BK / 2) * (BN / 2); i += 256) {  // 1024
            int kp = i >> 6;           // 0..15 (k pair)
            int c2 = i & 63;           // 0..63 (col pair)
            const uint32_t* gp =
                (const uint32_t*)(Bw + (size_t)(kk + kp * 2) * N + n_blk);
            uint32_t u0 = gp[c2];               // (k,   2c2), (k,   2c2+1)
            uint32_t u1 = gp[c2 + (N >> 1)];    // (k+1, 2c2), (k+1, 2c2+1)
            bt[(2 * c2)     * (LDB_T / 2) + kp] = (u0 & 0xffffu) | (u1 << 16);
            bt[(2 * c2 + 1) * (LDB_T / 2) + kp] = (u0 >> 16) | (u1 & 0xffff0000u);
        }
    };

    int buf = 0;
    stageAB(0, 0);
    wait_async0();
    __syncthreads();

    for (int k0 = 0; k0 < Kd; k0 += BK) {
        // Fragments from current buffer (128-bit LDS reads)
        v16bf afr[2], bfr[4];
        const int akb = half * 8;
        #pragma unroll
        for (int i2 = 0; i2 < 2; ++i2)
            afr[i2] = ld_frag_a(&As[buf][(m_wave + i2 * 16 + cn) * LDA_S + akb]);
        const int bkb = half * 8;
        #pragma unroll
        for (int t = 0; t < 4; ++t)
            bfr[t] = ld_frag_a(&Bt[buf][(n_wave + t * 16 + cn) * LDB_T + bkb]);

        // Prefetch next tile into the other buffer while we compute
        if (k0 + BK < Kd) stageAB(buf ^ 1, k0 + BK);

        #pragma unroll
        for (int i2 = 0; i2 < 2; ++i2)
            #pragma unroll
            for (int t = 0; t < 4; ++t)
                acc[i2][t] = __builtin_amdgcn_wmma_f32_16x16x32_bf16(
                    false, afr[i2], false, bfr[t], (short)0, acc[i2][t], false, false);

        wait_async0();
        __syncthreads();
        buf ^= 1;
    }

    // Epilogue: fused rank-4 LoRA up (+ scale) (+ bias); hoisted, branchless
    float upv[4][4];
    float bv[4];
    #pragma unroll
    for (int t = 0; t < 4; ++t) {
        int col = n_blk + n_wave + t * 16 + cn;
        #pragma unroll
        for (int j = 0; j < 4; ++j) upv[t][j] = lora_up[j * N + col];
        if (HAS_BIAS) bv[t] = bias[col];
    }
    #pragma unroll
    for (int i2 = 0; i2 < 2; ++i2) {
        #pragma unroll
        for (int r = 0; r < 8; ++r) {
            int row = m_blk + m_wave + i2 * 16 + half * 8 + r;
            const float4 tp = *(const float4*)(lora_t + (size_t)row * 4);
            #pragma unroll
            for (int t = 0; t < 4; ++t) {
                int col = n_blk + n_wave + t * 16 + cn;
                float v = acc[i2][t][r]
                        + tp.x * upv[t][0] + tp.y * upv[t][1]
                        + tp.z * upv[t][2] + tp.w * upv[t][3];
                if (SCALED) v *= out_scale;
                if (HAS_BIAS) v += bv[t];
                if (OUT_F32) out_f[(size_t)row * N + col] = v;
                else         out_bf[(size_t)row * N + col] = (bf16)v;
            }
        }
    }
}

// ---------------------------------------------------------------------------
// Flash attention, bf16 WMMA, fp32 softmax.  Q arrives pre-scaled by 0.125.
// Grid: (SQ/128, B*H). Block: 256 threads = 8 waves; wave owns 16 query rows.
// Double-buffered K/V staging: K via async DMA (row-major), V transposed for
// contiguous PV B-fragments; P staged through wave-private LDS.
// ---------------------------------------------------------------------------
#define BQ   128
#define BKV  64
#define KSTR 72      // halves per staged row/col (64 data + 8 pad) = 144B

__global__ __launch_bounds__(256)
void attn_wmma(const bf16* __restrict__ Q, const bf16* __restrict__ Kc,
               const bf16* __restrict__ V, bf16* __restrict__ O)
{
    __shared__ bf16 Kl[2][BKV * KSTR];       // [key][hd]
    __shared__ bf16 Vt[2][HDIM * KSTR];      // [hd col][key]  (transposed)
    __shared__ bf16 Pl[8][16][KSTR];         // per-wave P staging [row][key]

    const int tid  = threadIdx.x;
    const int lane = tid & 31;
    const int wave = tid >> 5;
    const int half = lane >> 4;
    const int cn   = lane & 15;

    const int bh = blockIdx.y;
    const int b  = bh / HH;
    const int h  = bh % HH;
    const int q0 = blockIdx.x * BQ;
    const int qw = q0 + wave * 16;           // wave's query rows

    // Q A-fragments (hd=64 -> 2 k-steps), 128-bit global reads
    v16bf qfr[2];
    {
        const bf16* qp = Q + ((size_t)(b * SQL) + qw + cn) * DD + h * HDIM;
        #pragma unroll
        for (int ks = 0; ks < 2; ++ks)
            qfr[ks] = ld_frag_a(qp + ks * 32 + half * 8);
    }

    float mrow[8], lrow[8];
    v8f oacc[4];
    #pragma unroll
    for (int r = 0; r < 8; ++r) { mrow[r] = -1e30f; lrow[r] = 0.f; }
    #pragma unroll
    for (int t = 0; t < 4; ++t)
        #pragma unroll
        for (int j = 0; j < 8; ++j) oacc[t][j] = 0.f;

    // Stage K/V chunk at kvbase into buffer sb
    auto stageKV = [&](int sb, int kvbase) {
        uint64_t kb64 = (uint64_t)(uintptr_t)
            (Kc + ((size_t)(b * SKVL) + kvbase) * DD + h * HDIM);
        uint32_t klB = lds_off(&Kl[sb][0]);
        for (int i = tid; i < BKV * 8; i += 256) {      // 512 granules
            uint32_t row = (uint32_t)i >> 3, g = (uint32_t)i & 7;
            async_b128(klB + row * (KSTR * 2) + g * 16,
                       row * (uint32_t)(DD * 2) + g * 16, kb64);
        }
        uint32_t* vt = (uint32_t*)&Vt[sb][0];
        const uint32_t* gvb = (const uint32_t*)
            (V + ((size_t)(b * SKVL) + kvbase) * DD + h * HDIM);
        for (int i = tid; i < (BKV / 2) * (HDIM / 2); i += 256) {  // 1024
            int kp = i >> 5;          // 0..31 (key pair)
            int c2 = i & 31;          // 0..31 (hd col pair)
            const uint32_t* gp = gvb + (size_t)(kp * 2) * (DD / 2);
            uint32_t u0 = gp[c2];
            uint32_t u1 = gp[c2 + (DD / 2)];
            vt[(2 * c2)     * (KSTR / 2) + kp] = (u0 & 0xffffu) | (u1 << 16);
            vt[(2 * c2 + 1) * (KSTR / 2) + kp] = (u0 >> 16) | (u1 & 0xffff0000u);
        }
    };

    int buf = 0;
    stageKV(0, 0);
    wait_async0();
    __syncthreads();

    const int NCH = SKVL / BKV;
    for (int c = 0; c < NCH; ++c) {
        // Prefetch next chunk into the other buffer
        if (c + 1 < NCH) stageKV(buf ^ 1, (c + 1) * BKV);

        const bf16* KlC = &Kl[buf][0];
        const bf16* VtC = &Vt[buf][0];

        // S = (Q*0.125) . K^T  (4 key tiles of 16, 2 k-steps over hd)
        v8f s[4];
        #pragma unroll
        for (int t = 0; t < 4; ++t) {
            #pragma unroll
            for (int j = 0; j < 8; ++j) s[t][j] = 0.f;
            const bf16* kp = &KlC[(t * 16 + cn) * KSTR];
            #pragma unroll
            for (int ks = 0; ks < 2; ++ks) {
                v16bf kf = ld_frag_a(kp + ks * 32 + half * 8);
                s[t] = __builtin_amdgcn_wmma_f32_16x16x32_bf16(
                    false, qfr[ks], false, kf, (short)0, s[t], false, false);
            }
        }

        // Online softmax (rows split across lane halves; reduce within 16 lanes)
        float mnew[8], alpha[8], csum[8];
        #pragma unroll
        for (int r = 0; r < 8; ++r) {
            float cm = s[0][r];
            #pragma unroll
            for (int t = 1; t < 4; ++t) cm = fmaxf(cm, s[t][r]);
            #pragma unroll
            for (int msk = 1; msk < 16; msk <<= 1)
                cm = fmaxf(cm, __shfl_xor(cm, msk, 32));
            mnew[r]  = fmaxf(mrow[r], cm);
            alpha[r] = __expf(mrow[r] - mnew[r]);
            mrow[r]  = mnew[r];
            csum[r]  = 0.f;
        }
        // P = exp(S - m) -> wave-private LDS (same-wave LDS ops are in-order)
        #pragma unroll
        for (int t = 0; t < 4; ++t) {
            #pragma unroll
            for (int r = 0; r < 8; ++r) {
                float p = __expf(s[t][r] - mnew[r]);
                csum[r] += p;
                Pl[wave][half * 8 + r][t * 16 + cn] = (bf16)p;
            }
        }
        #pragma unroll
        for (int r = 0; r < 8; ++r) {
            float cs = csum[r];
            #pragma unroll
            for (int msk = 1; msk < 16; msk <<= 1)
                cs += __shfl_xor(cs, msk, 32);
            lrow[r] = lrow[r] * alpha[r] + cs;
        }
        // Rescale existing accumulators
        #pragma unroll
        for (int t = 0; t < 4; ++t)
            #pragma unroll
            for (int r = 0; r < 8; ++r) oacc[t][r] *= alpha[r];

        // P A-fragments from wave-private LDS (128-bit reads)
        v16bf pf[2];
        #pragma unroll
        for (int ks = 0; ks < 2; ++ks)
            pf[ks] = ld_frag_a(&Pl[wave][cn][ks * 32 + half * 8]);

        // O += P . V  (V fragments contiguous from transposed tile)
        #pragma unroll
        for (int t2 = 0; t2 < 4; ++t2) {
            const bf16* vp = &VtC[(t2 * 16 + cn) * KSTR];
            #pragma unroll
            for (int ks = 0; ks < 2; ++ks) {
                v16bf vf = ld_frag_a(vp + ks * 32 + half * 8);
                oacc[t2] = __builtin_amdgcn_wmma_f32_16x16x32_bf16(
                    false, pf[ks], false, vf, (short)0, oacc[t2], false, false);
            }
        }

        wait_async0();
        __syncthreads();
        buf ^= 1;
    }

    // Normalize and store bf16 attention output (B,SQ,H,hd) == (M, D)
    #pragma unroll
    for (int t2 = 0; t2 < 4; ++t2) {
        #pragma unroll
        for (int r = 0; r < 8; ++r) {
            int row = qw + half * 8 + r;
            int col = h * HDIM + t2 * 16 + cn;
            O[((size_t)(b * SQL) + row) * DD + col] = (bf16)(oacc[t2][r] / lrow[r]);
        }
    }
}

// ---------------------------------------------------------------------------
// Host launch
// ---------------------------------------------------------------------------
extern "C" void kernel_launch(void* const* d_in, const int* in_sizes, int n_in,
                              void* d_out, int out_size, void* d_ws, size_t ws_size,
                              hipStream_t stream)
{
    const float* hs   = (const float*)d_in[0];
    const float* enc  = (const float*)d_in[1];
    const float* Wq   = (const float*)d_in[2];
    const float* Wk   = (const float*)d_in[3];
    const float* Wv   = (const float*)d_in[4];
    const float* Wo   = (const float*)d_in[5];
    const float* bo   = (const float*)d_in[6];
    const float* q_dn = (const float*)d_in[7];
    const float* q_up = (const float*)d_in[8];
    const float* k_dn = (const float*)d_in[9];
    const float* k_up = (const float*)d_in[10];
    const float* v_dn = (const float*)d_in[11];
    const float* v_up = (const float*)d_in[12];
    const float* o_dn = (const float*)d_in[13];
    const float* o_up = (const float*)d_in[14];

    const int M = Bb * SQL;          // 8192
    const size_t actElems = (size_t)M * DD;       // 8.4M
    const size_t wElems   = (size_t)DD * DD;      // 1M

    char* w = (char*)d_ws;
    size_t off = 0;
    auto carve = [&](size_t bytes) -> void* {
        void* p = w + off;
        off += (bytes + 255) & ~(size_t)255;
        return p;
    };
    bf16* Xb  = (bf16*)carve(actElems * 2);
    bf16* Eb  = (bf16*)carve(actElems * 2);
    bf16* Wqb = (bf16*)carve(wElems * 2);
    bf16* Wkb = (bf16*)carve(wElems * 2);
    bf16* Wvb = (bf16*)carve(wElems * 2);
    bf16* Wob = (bf16*)carve(wElems * 2);
    bf16* Qb  = (bf16*)carve(actElems * 2);
    bf16* Kb  = (bf16*)carve(actElems * 2);
    bf16* Vb  = (bf16*)carve(actElems * 2);
    bf16* Ob  = (bf16*)carve(actElems * 2);
    float* tq = (float*)carve((size_t)M * 4 * 4);
    float* tk = (float*)carve((size_t)M * 4 * 4);
    float* tv = (float*)carve((size_t)M * 4 * 4);
    float* to = (float*)carve((size_t)M * 4 * 4);

    auto cvt = [&](const float* in, bf16* out, size_t n) {
        int blocks = (int)((n / 4 + 255) / 256);
        cvt_f32_bf16<<<blocks, 256, 0, stream>>>(in, out, (int)n);
    };
    cvt(hs,  Xb,  actElems);
    cvt(enc, Eb,  actElems);
    cvt(Wq,  Wqb, wElems);
    cvt(Wk,  Wkb, wElems);
    cvt(Wv,  Wvb, wElems);
    cvt(Wo,  Wob, wElems);

    dim3 ldGrid((M * 4 + 255) / 256);
    lora_down_kernel<<<ldGrid, 256, 0, stream>>>(Xb, q_dn, tq, M, DD);
    lora_down_kernel<<<ldGrid, 256, 0, stream>>>(Eb, k_dn, tk, M, DD);
    lora_down_kernel<<<ldGrid, 256, 0, stream>>>(Eb, v_dn, tv, M, DD);

    dim3 gGrid(DD / BN, M / BM);   // (8, 64)
    // Q projection folds softmax scale 1/sqrt(64)=0.125 (exact in bf16)
    gemm_bf16_wmma<false, false, true><<<gGrid, 256, 0, stream>>>(
        Xb, Wqb, tq, q_up, nullptr, Qb, nullptr, 0.125f, M, DD, DD);
    gemm_bf16_wmma<false, false, false><<<gGrid, 256, 0, stream>>>(
        Eb, Wkb, tk, k_up, nullptr, Kb, nullptr, 1.0f, M, DD, DD);
    gemm_bf16_wmma<false, false, false><<<gGrid, 256, 0, stream>>>(
        Eb, Wvb, tv, v_up, nullptr, Vb, nullptr, 1.0f, M, DD, DD);

    dim3 aGrid(SQL / BQ, Bb * HH); // (16, 64)
    attn_wmma<<<aGrid, 256, 0, stream>>>(Qb, Kb, Vb, Ob);

    lora_down_kernel<<<ldGrid, 256, 0, stream>>>(Ob, o_dn, to, M, DD);
    gemm_bf16_wmma<true, true, false><<<gGrid, 256, 0, stream>>>(
        Ob, Wob, to, o_up, bo, nullptr, (float*)d_out, 1.0f, M, DD, DD);
}
